// SelfAttentionBlock_64132451664383
// MI455X (gfx1250) — matrix operand order repaired and checked by reading, not compiled, not run on hardware
//
#include <hip/hip_runtime.h>
#include <hip/hip_bf16.h>
#include <math.h>

typedef __attribute__((ext_vector_type(16))) _Float16 v16h;
typedef __attribute__((ext_vector_type(8)))  _Float16 v8h;
typedef __attribute__((ext_vector_type(8)))  float    v8f;

#define B_   8
#define C_   512
#define HD   64
#define N_   4096

// ---------------------------------------------------------------------------
// Kernel 1: 1x1-conv projections q,k,v with bias, f32 -> f16, writing the
// layouts the WMMA attention kernel wants:
//   qT [B][N][64]   (row = query position, contiguous head-dim -> A operand)
//   kT [B][N][64]   (row = key   position, contiguous head-dim -> B operand)
//   v16[B][512][N]  (row = channel, contiguous position       -> B operand)
// One block per (b, o, n-tile of 1024). Weight row w[o][:] is block-uniform
// (scalar loads); x reads are lane-coalesced and L2-resident (x = 67MB < 192MB).
// ---------------------------------------------------------------------------
__global__ __launch_bounds__(128)
void proj_kernel(const float* __restrict__ x,
                 const float* __restrict__ wq, const float* __restrict__ bq,
                 const float* __restrict__ wk, const float* __restrict__ bk,
                 const float* __restrict__ wv, const float* __restrict__ bv,
                 _Float16* __restrict__ qT,
                 _Float16* __restrict__ kT,
                 _Float16* __restrict__ v16)
{
    const int tid = threadIdx.x;           // 0..127
    const int n0  = blockIdx.x * 1024;     // n tile base
    const int o   = blockIdx.y;            // 0..639 (64 q, 64 k, 512 v)
    const int b   = blockIdx.z;

    const float* w;
    float bias;
    if (o < 64)        { w = wq + (size_t)o * C_;         bias = bq[o];        }
    else if (o < 128)  { w = wk + (size_t)(o - 64) * C_;  bias = bk[o - 64];   }
    else               { w = wv + (size_t)(o - 128) * C_; bias = bv[o - 128];  }

    float acc[8];
#pragma unroll
    for (int j = 0; j < 8; ++j) acc[j] = 0.f;

    const float* xb = x + (size_t)b * C_ * N_ + n0 + tid;
    for (int c = 0; c < C_; ++c) {
        const float wc = w[c];                       // wave-uniform (scalar)
        const float* xr = xb + (size_t)c * N_;
#pragma unroll
        for (int j = 0; j < 8; ++j)
            acc[j] = fmaf(wc, xr[j * 128], acc[j]);  // lanes coalesced
    }

    if (o < 64) {
#pragma unroll
        for (int j = 0; j < 8; ++j) {
            const int n = n0 + j * 128 + tid;
            qT[((size_t)(b * N_ + n)) * HD + o] = (_Float16)(acc[j] + bias);
        }
    } else if (o < 128) {
        const int oo = o - 64;
#pragma unroll
        for (int j = 0; j < 8; ++j) {
            const int n = n0 + j * 128 + tid;
            kT[((size_t)(b * N_ + n)) * HD + oo] = (_Float16)(acc[j] + bias);
        }
    } else {
        const int oo = o - 128;
#pragma unroll
        for (int j = 0; j < 8; ++j) {
            const int n = n0 + j * 128 + tid;
            v16[((size_t)(b * C_ + oo)) * N_ + n] = (_Float16)(acc[j] + bias);
        }
    }
}

// ---------------------------------------------------------------------------
// Kernel 2: fused flash-attention, f16 WMMA, f32 accumulation.
// Block = 256 threads = 8 waves: 2 row-groups (16 query rows each) x 4
// channel-groups (128 output channels each). Per 64-wide key tile:
//   cg==0 wave : S = Q^T K (8x v_wmma_f32_16x16x32_f16), online softmax,
//                P(f16) + rescale factors -> LDS
//   all waves  : rescale O accumulators, O += P * V^T (16 WMMAs)
// K/V fragments are gathered directly from global (fully L2-resident);
// the next tile's lines are pulled near with global_prefetch_b8.
// ---------------------------------------------------------------------------
__global__ __launch_bounds__(256)
void attn_kernel(const _Float16* __restrict__ qT,
                 const _Float16* __restrict__ kT,
                 const _Float16* __restrict__ v16,
                 const float*    __restrict__ x,
                 const float*    __restrict__ coeff,
                 float*          __restrict__ out)
{
    __shared__ alignas(32) _Float16 p16[2][16][64];  // softmax weights per rg
    __shared__ float scaleL[2][16];                  // exp(m_old - m_new)
    __shared__ float lsumL[2][16];                   // final row sums

    const int tid  = threadIdx.x;
    const int wid  = tid >> 5;         // wave 0..7
    const int lane = tid & 31;
    const int rg   = wid >> 2;         // row group 0..1
    const int cg   = wid & 3;          // channel group 0..3
    const int b    = blockIdx.y;
    const int nq0  = blockIdx.x * 32 + rg * 16;

    const int lrow = lane & 15;        // row (A) / column (B,D) index
    const int lhi  = lane >> 4;        // half-wave select
    const int abase = lhi * 16;        // A-fragment K chunk
    const int kb    = lhi * 16;        // B-fragment K chunk

    // O accumulators: 8 fragments cover channels cg*128 .. cg*128+127
    v8f acc[8];
#pragma unroll
    for (int g = 0; g < 8; ++g)
#pragma unroll
        for (int v = 0; v < 8; ++v) acc[g][v] = 0.f;

    // online softmax state (lanes keep redundant copies per 16-lane half)
    float m_i[8], l_i[8];
#pragma unroll
    for (int v = 0; v < 8; ++v) { m_i[v] = -INFINITY; l_i[v] = 0.f; }

    // Q A-fragments (16 rows x 64 head-dim -> two 16x32 fragments), kept
    // in registers for the whole key loop.
    v16h aq[2];
    {
        const _Float16* qrow = qT + ((size_t)(b * N_ + nq0 + lrow)) * HD;
#pragma unroll
        for (int h = 0; h < 2; ++h) {
            v8h lo = *(const v8h*)(qrow + h * 32 + abase);
            v8h hi = *(const v8h*)(qrow + h * 32 + abase + 8);
#pragma unroll
            for (int j = 0; j < 8; ++j) { aq[h][j] = lo[j]; aq[h][8 + j] = hi[j]; }
        }
    }

    const int is_smax = __builtin_amdgcn_readfirstlane(cg == 0 ? 1 : 0);

    for (int m0 = 0; m0 < N_; m0 += 64) {
        // ---- prefetch next key tile (K rows + this wave's V rows) ----
        const int mN = m0 + 64;
        if (mN < N_) {
            // 64 K rows of 128B each: lanes cover rows 0..31 and 32..63
            __builtin_prefetch(kT + ((size_t)(b * N_ + mN + lane)) * HD, 0, 3);
            __builtin_prefetch(kT + ((size_t)(b * N_ + mN + 32 + lane)) * HD, 0, 3);
            // 128 V rows (this wave's channels), 128B of keys each
#pragma unroll
            for (int i = 0; i < 4; ++i)
                __builtin_prefetch(
                    v16 + ((size_t)(b * C_ + cg * 128 + i * 32 + lane)) * N_ + mN,
                    0, 3);
        }

        if (is_smax) {
            // ---- S = Q^T K over a 16x64 tile (4 column sub-tiles) ----
            v8f sfr[4];
#pragma unroll
            for (int t = 0; t < 4; ++t) {
                const _Float16* krow =
                    kT + ((size_t)(b * N_ + m0 + t * 16 + lrow)) * HD;
                v16h b0 = *(const v16h*)(krow + kb);        // head-dim  0..31
                v16h b1 = *(const v16h*)(krow + 32 + kb);   // head-dim 32..63
                v8f s;
#pragma unroll
                for (int v = 0; v < 8; ++v) s[v] = 0.f;
                s = __builtin_amdgcn_wmma_f32_16x16x32_f16(
                        false, aq[0], false, b0, (short)0, s, false, false);
                s = __builtin_amdgcn_wmma_f32_16x16x32_f16(
                        false, aq[1], false, b1, (short)0, s, false, false);
                sfr[t] = s;
            }
            // ---- online softmax (row = v + 8*lhi, cols across 16 lanes) ----
#pragma unroll
            for (int v = 0; v < 8; ++v) {
                float mx = fmaxf(fmaxf(sfr[0][v], sfr[1][v]),
                                 fmaxf(sfr[2][v], sfr[3][v]));
#pragma unroll
                for (int d = 1; d < 16; d <<= 1)
                    mx = fmaxf(mx, __shfl_xor(mx, d, 32));
                const float mnew = fmaxf(m_i[v], mx);
                const float sc   = __expf(m_i[v] - mnew);   // 0 on first tile
                const int   row  = v + 8 * lhi;
                float rs = 0.f;
#pragma unroll
                for (int t = 0; t < 4; ++t) {
                    const float p = __expf(sfr[t][v] - mnew);
                    rs += p;
                    p16[rg][row][t * 16 + lrow] = (_Float16)p;
                }
#pragma unroll
                for (int d = 1; d < 16; d <<= 1)
                    rs += __shfl_xor(rs, d, 32);
                l_i[v] = l_i[v] * sc + rs;
                m_i[v] = mnew;
                if (lrow == 0) scaleL[rg][row] = sc;
            }
        }
        __syncthreads();

        // ---- rescale O accumulators by exp(m_old - m_new) per row ----
        float sc[8];
#pragma unroll
        for (int v = 0; v < 8; ++v) sc[v] = scaleL[rg][v + 8 * lhi];
#pragma unroll
        for (int g = 0; g < 8; ++g)
#pragma unroll
            for (int v = 0; v < 8; ++v) acc[g][v] *= sc[v];

        // ---- P A-fragments (16 x 64 -> two 16x32 fragments) from LDS ----
        v16h ap[2];
        {
            const _Float16* prow = &p16[rg][lrow][0];
#pragma unroll
            for (int h = 0; h < 2; ++h) {
                v8h lo = *(const v8h*)(prow + h * 32 + abase);
                v8h hi = *(const v8h*)(prow + h * 32 + abase + 8);
#pragma unroll
                for (int j = 0; j < 8; ++j) { ap[h][j] = lo[j]; ap[h][8 + j] = hi[j]; }
            }
        }

        // ---- O += P * V^T for this wave's 128 channels ----
#pragma unroll
        for (int g = 0; g < 8; ++g) {
            const _Float16* vrow =
                v16 + ((size_t)(b * C_ + cg * 128 + g * 16 + lrow)) * N_ + m0;
            v16h b0 = *(const v16h*)(vrow + kb);        // keys m0+ 0..31
            v16h b1 = *(const v16h*)(vrow + 32 + kb);   // keys m0+32..63
            acc[g] = __builtin_amdgcn_wmma_f32_16x16x32_f16(
                         false, ap[0], false, b0, (short)0, acc[g], false, false);
            acc[g] = __builtin_amdgcn_wmma_f32_16x16x32_f16(
                         false, ap[1], false, b1, (short)0, acc[g], false, false);
        }
        __syncthreads();
    }

    // ---- publish final row sums ----
    if (is_smax && lrow == 0) {
#pragma unroll
        for (int v = 0; v < 8; ++v) lsumL[rg][v + 8 * lhi] = l_i[v];
    }
    __syncthreads();

    // ---- epilogue: out = coeff * (O / l) + x ----
    const float cf = coeff[0];
    float invl[8];
#pragma unroll
    for (int v = 0; v < 8; ++v) invl[v] = 1.f / lsumL[rg][v + 8 * lhi];

#pragma unroll
    for (int g = 0; g < 8; ++g) {
        const int c = cg * 128 + g * 16 + lrow;
#pragma unroll
        for (int v = 0; v < 8; ++v) {
            const int n = nq0 + v + 8 * lhi;
            const size_t idx = ((size_t)(b * C_ + c)) * N_ + n;
            out[idx] = cf * (acc[g][v] * invl[v]) + x[idx];
        }
    }
}

// ---------------------------------------------------------------------------
extern "C" void kernel_launch(void* const* d_in, const int* in_sizes, int n_in,
                              void* d_out, int out_size, void* d_ws, size_t ws_size,
                              hipStream_t stream)
{
    const float* x     = (const float*)d_in[0];
    const float* wq    = (const float*)d_in[1];
    const float* bq    = (const float*)d_in[2];
    const float* wk    = (const float*)d_in[3];
    const float* bk    = (const float*)d_in[4];
    const float* wv    = (const float*)d_in[5];
    const float* bv    = (const float*)d_in[6];
    const float* coeff = (const float*)d_in[7];
    float* out = (float*)d_out;

    // Workspace: qT (4MB) | kT (4MB) | v16 (33.5MB)  == 41.9 MB of f16
    _Float16* qT  = (_Float16*)d_ws;
    _Float16* kT  = qT + (size_t)B_ * N_ * HD;
    _Float16* v16 = kT + (size_t)B_ * N_ * HD;

    dim3 g1(N_ / 1024, 640, B_);   // (n-tiles, q+k+v output rows, batch)
    proj_kernel<<<g1, 128, 0, stream>>>(x, wq, bq, wk, bk, wv, bv, qT, kT, v16);

    dim3 g2(N_ / 32, B_);          // 32 query rows per block
    attn_kernel<<<g2, 256, 0, stream>>>(qT, kT, v16, x, coeff, out);
}
